// ComplexBlockLinear_9818295238731
// MI455X (gfx1250) — matrix operand order repaired and checked by reading, not compiled, not run on hardware
//
#include <hip/hip_runtime.h>

typedef _Float16 v16h __attribute__((ext_vector_type(16)));
typedef float v8f __attribute__((ext_vector_type(8)));
typedef unsigned int u32x4 __attribute__((ext_vector_type(4)));
typedef int i32x4 __attribute__((ext_vector_type(4)));
typedef int i32x8 __attribute__((ext_vector_type(8)));

#define NBLK 8
#define BS 128
#define HDIM 1024
#define MROWS 32768                  // B*S
#define ROWSTR 1040                  // 1024 data + 16 pad floats (4 dwords per 256)
#define XI_OFF (16 * ROWSTR)
#define WPK_PER_BLK (6 * 8 * 4 * 32 * 16)   // 98304 halves per block n
#define WPK_TOTAL (NBLK * WPK_PER_BLK)      // 786432 halves = 1.5 MB

// ---------------- weight prep: pack B fragments (hi/lo split, + negated wi) ----
// layout: [n][mat(6: wr_h,wr_l,wi_h,wi_l,-wi_h,-wi_l)][ntile(8)][kstep(4)][lane(32)][half(16)]
// B-operand VGPR layout used here (contiguous K, same order as A): lane&15 = N column, lane>>4 picks K half:
//   half j -> K = kstep*32 + 16*(lane>>4) + j
__global__ void cbl_prep(const float* __restrict__ w, _Float16* __restrict__ pk) {
    int idx = blockIdx.x * 256 + threadIdx.x;
    if (idx >= WPK_TOTAL) return;
    int j    = idx & 15;
    int lane = (idx >> 4) & 31;
    int kk   = (idx >> 9) & 3;
    int nt   = (idx >> 11) & 7;
    int mat  = (idx >> 14) % 6;
    int n    = idx / WPK_PER_BLK;
    int K = kk * 32 + ((lane >> 4) << 4) + j;
    int N = nt * 16 + (lane & 15);
    int c = (mat < 2) ? 0 : 1;                       // 0 = re, 1 = im plane of weight
    float sgn = (mat >= 4) ? -1.0f : 1.0f;           // negated wi copies
    float val = sgn * w[(((n * BS + K) * BS + N) << 1) + c];
    _Float16 hi = (_Float16)val;
    pk[idx] = (mat & 1) ? (_Float16)(val - (float)hi) : hi;
}

__device__ __forceinline__ v8f wmma_f16(v16h a, v16h b, v8f c) {
    return __builtin_amdgcn_wmma_f32_16x16x32_f16(false, a, false, b, (short)0, c,
                                                  false, false);
}

// ---------------- main kernel: 2048 WGs x 256 threads -------------------------
// WG = 16 contiguous rows; wave w computes block n = w (16 rows x 128 cols, re+im)
__global__ void __launch_bounds__(256) cbl_main(const float* __restrict__ xr,
                                                const float* __restrict__ xi,
                                                const _Float16* __restrict__ wpk,
                                                float2* __restrict__ out) {
    __shared__ __align__(16) float smem[2 * XI_OFF];   // 133 KB of 320 KB LDS
    const int tid  = threadIdx.x;
    const int wave = tid >> 5;          // block n
    const int lane = tid & 31;
    const int m    = lane & 15;
    const int t    = lane >> 4;
    const long r0  = (long)blockIdx.x * 16;

#if __has_builtin(__builtin_amdgcn_tensor_load_to_lds)
    if (tid == 0) {
        unsigned lds0 = (unsigned)(size_t)&smem[0];
        unsigned lds1 = (unsigned)(size_t)&smem[XI_OFF];
        unsigned long long ga0 = (unsigned long long)(xr + r0 * HDIM);
        unsigned long long ga1 = (unsigned long long)(xi + r0 * HDIM);
        // group1: data_size=4B, pad_enable, pad_interval=256 dwords, pad_amount=4 dwords,
        // tensor_dim0=1024, tensor_dim1=16, tile_dim0=1024, tile_dim1=16, dim0_stride=1024
        unsigned dw0 = (2u << 16) | (1u << 20) | (7u << 22) | (3u << 25);
        i32x8 g1 = { (int)dw0, (int)(1024u << 16), (int)(16u << 16), (int)(1024u << 16),
                     16, 1024, 0, 0 };
        u32x4 g0a = { 1u, lds0, (unsigned)ga0,
                      ((unsigned)((ga0 >> 32) & 0x1FFFFFFu)) | 0x80000000u };
        u32x4 g0b = { 1u, lds1, (unsigned)ga1,
                      ((unsigned)((ga1 >> 32) & 0x1FFFFFFu)) | 0x80000000u };
        i32x4 z4 = { 0, 0, 0, 0 };
#if __clang_major__ >= 23
        i32x8 z8 = { 0, 0, 0, 0, 0, 0, 0, 0 };
        __builtin_amdgcn_tensor_load_to_lds(g0a, g1, z4, z4, z8, 0);
        __builtin_amdgcn_tensor_load_to_lds(g0b, g1, z4, z4, z8, 0);
#else
        __builtin_amdgcn_tensor_load_to_lds(g0a, g1, z4, z4, 0);
        __builtin_amdgcn_tensor_load_to_lds(g0b, g1, z4, z4, 0);
#endif
        __builtin_amdgcn_s_wait_tensorcnt(0);
    }
#else
    {   // fallback staging with identical padded layout
        const float* srcr = xr + r0 * HDIM;
        const float* srci = xi + r0 * HDIM;
        for (int i = tid; i < 16 * HDIM / 4; i += 256) {
            int K = (i * 4) & (HDIM - 1);
            int r = (i * 4) >> 10;
            int di = r * ROWSTR + K + ((K >> 8) << 2);
            *(float4*)&smem[di]          = *(const float4*)(srcr + i * 4);
            *(float4*)&smem[XI_OFF + di] = *(const float4*)(srci + i * 4);
        }
    }
#endif
    __syncthreads();

    // ---- A fragments: convert f32 -> f16 hi/lo once (A layout: lane&15 = row,
    //      halves 0..7 -> K = t*16 + j, halves 8..15 -> K = t*16 + 8 + j) ----
    v16h a_rh[4], a_rl[4], a_ih[4], a_il[4];
#pragma unroll
    for (int kk = 0; kk < 4; ++kk) {
        const int K0 = wave * BS + kk * 32 + t * 16;
        const int K1 = K0 + 8;
        const int i0 = m * ROWSTR + K0 + ((K0 >> 8) << 2);
        const int i1 = m * ROWSTR + K1 + ((K1 >> 8) << 2);
        float f[16];
        *(float4*)&f[0]  = *(const float4*)&smem[i0];
        *(float4*)&f[4]  = *(const float4*)&smem[i0 + 4];
        *(float4*)&f[8]  = *(const float4*)&smem[i1];
        *(float4*)&f[12] = *(const float4*)&smem[i1 + 4];
#pragma unroll
        for (int q = 0; q < 16; ++q) {
            _Float16 h = (_Float16)f[q];
            a_rh[kk][q] = h;
            a_rl[kk][q] = (_Float16)(f[q] - (float)h);
        }
        *(float4*)&f[0]  = *(const float4*)&smem[XI_OFF + i0];
        *(float4*)&f[4]  = *(const float4*)&smem[XI_OFF + i0 + 4];
        *(float4*)&f[8]  = *(const float4*)&smem[XI_OFF + i1];
        *(float4*)&f[12] = *(const float4*)&smem[XI_OFF + i1 + 4];
#pragma unroll
        for (int q = 0; q < 16; ++q) {
            _Float16 h = (_Float16)f[q];
            a_ih[kk][q] = h;
            a_il[kk][q] = (_Float16)(f[q] - (float)h);
        }
    }

    // ---- main loop: 8 N-tiles, 4 k-steps, 12 WMMAs each, 4 parallel chains ----
    const _Float16* wb = wpk + (size_t)wave * WPK_PER_BLK + (size_t)lane * 16;
    for (int nt = 0; nt < 8; ++nt) {
        v8f re0 = { 0, 0, 0, 0, 0, 0, 0, 0 };
        v8f re1 = { 0, 0, 0, 0, 0, 0, 0, 0 };
        v8f im0 = { 0, 0, 0, 0, 0, 0, 0, 0 };
        v8f im1 = { 0, 0, 0, 0, 0, 0, 0, 0 };
#pragma unroll
        for (int kk = 0; kk < 4; ++kk) {
            const _Float16* fb = wb + (nt * 4 + kk) * 512;
            v16h brh = *(const v16h*)(fb + 0 * 16384);
            v16h brl = *(const v16h*)(fb + 1 * 16384);
            v16h bih = *(const v16h*)(fb + 2 * 16384);
            v16h bil = *(const v16h*)(fb + 3 * 16384);
            v16h bnh = *(const v16h*)(fb + 4 * 16384);   // -wi hi
            v16h bnl = *(const v16h*)(fb + 5 * 16384);   // -wi lo
            // out_re = xr*wr - xi*wi ; out_im = xr*wi + xi*wr (hh + hl + lh splits)
            // interleaved re0/im0/re1/im1 so same-accumulator WMMAs are 4 apart
            re0 = wmma_f16(a_rh[kk], brh, re0);
            im0 = wmma_f16(a_rh[kk], bih, im0);
            re1 = wmma_f16(a_rl[kk], brh, re1);
            im1 = wmma_f16(a_rl[kk], bih, im1);
            re0 = wmma_f16(a_rh[kk], brl, re0);
            im0 = wmma_f16(a_rh[kk], bil, im0);
            re1 = wmma_f16(a_ih[kk], bnl, re1);
            im1 = wmma_f16(a_ih[kk], brl, im1);
            re0 = wmma_f16(a_ih[kk], bnh, re0);
            im0 = wmma_f16(a_ih[kk], brh, im0);
            re1 = wmma_f16(a_il[kk], bnh, re1);
            im1 = wmma_f16(a_il[kk], brh, im1);
        }
        v8f acc_re = re0 + re1;
        v8f acc_im = im0 + im1;
        // C layout: lane&15 = N, M = (lane>>4)*8 + vgpr; pack (re,im) -> float2
        const int col = wave * BS + nt * 16 + m;
#pragma unroll
        for (int g = 0; g < 8; ++g) {
            out[(r0 + t * 8 + g) * HDIM + col] = make_float2(acc_re[g], acc_im[g]);
        }
    }
}

extern "C" void kernel_launch(void* const* d_in, const int* in_sizes, int n_in,
                              void* d_out, int out_size, void* d_ws, size_t ws_size,
                              hipStream_t stream) {
    const float* xr = (const float*)d_in[0];
    const float* xi = (const float*)d_in[1];
    const float* w  = (const float*)d_in[2];
    _Float16* wpk = (_Float16*)d_ws;   // needs 1.5 MB scratch
    cbl_prep<<<WPK_TOTAL / 256, 256, 0, stream>>>(w, wpk);
    cbl_main<<<MROWS / 16, 256, 0, stream>>>(xr, xi, wpk, (float2*)d_out);
}
